// LRU_57483842289812
// MI455X (gfx1250) — matrix operand order repaired and checked by reading, not compiled, not run on hardware
//
#include <hip/hip_runtime.h>
#include <hip/hip_bf16.h>

// ---------------- problem constants ----------------
#define T_DIM 2048
#define B_DIM 16
#define M_DIM 512
#define H_DIM 256
#define S_DIM (T_DIM * B_DIM)      // 32768 flattened (t,b) rows
#define CHUNK 32
#define NCHUNK (T_DIM / CHUNK)     // 64

typedef __bf16 bf16_t;
typedef __attribute__((ext_vector_type(16))) __bf16 v16bf;
typedef __attribute__((ext_vector_type(8)))  __bf16 v8bf;
typedef __attribute__((ext_vector_type(8)))  float  v8f;

// ---------------- WMMA fragment helpers (wave32, gfx1250) ----------------
// A-matrix 16x32 bf16 (contiguous k order chosen here; B uses the same): lane = hi*16 + r holds row M=r,
// halves j=0..15 -> K = hi*16 + j.
__device__ __forceinline__ v16bf load_a_frag(const bf16_t* A, int lda,
                                             int row0, int k0, int lane) {
    const int r = lane & 15, hi = lane >> 4;
    const bf16_t* p = A + (size_t)(row0 + r) * lda + k0 + hi * 16;
    v8bf lo  = *(const v8bf*)p;          // K = hi*16 .. hi*16+7
    v8bf hi8 = *(const v8bf*)(p + 8);    // K = hi*16+8 .. +7
    v16bf out;
#pragma unroll
    for (int j = 0; j < 8; ++j) { out[j] = lo[j]; out[8 + j] = hi8[j]; }
    return out;
}

// Combine two 16-B loads into a v16bf (assumes only 16-B alignment).
__device__ __forceinline__ v16bf ld16bf(const char* p) {
    v8bf lo  = *(const v8bf*)p;
    v8bf hi8 = *(const v8bf*)(p + 16);
    v16bf out;
#pragma unroll
    for (int j = 0; j < 8; ++j) { out[j] = lo[j]; out[8 + j] = hi8[j]; }
    return out;
}

__device__ __forceinline__ v8f wmma_bf16(v16bf a, v16bf b, v8f c) {
    return __builtin_amdgcn_wmma_f32_16x16x32_bf16(
        /*neg_a=*/false, a, /*neg_b=*/false, b,
        /*c_mod=*/(short)0, c, /*reuse_a=*/false, /*reuse_b=*/false);
}

// C/D 16x16 f32: VGPR i -> M = hi*8 + i, N = lane&15.
__device__ __forceinline__ void store_tile(float* C, int ldc, int row0, int col0,
                                           int lane, v8f acc) {
    const int n = lane & 15, hi = lane >> 4;
#pragma unroll
    for (int i = 0; i < 8; ++i)
        C[(size_t)(row0 + hi * 8 + i) * ldc + col0 + n] = acc[i];
}

// Async global->LDS stage: one 16-B element per lane (512 B per wave).
__device__ __forceinline__ void async_stage16(unsigned ldsAddr, const bf16_t* gsrc) {
    asm volatile("global_load_async_to_lds_b128 %0, %1, off"
                 :: "v"(ldsAddr), "v"(gsrc) : "memory");
}

// ---------------- prep kernels ----------------
__global__ void lru_cvt_x(const float* __restrict__ X, bf16_t* __restrict__ Xb) {
    size_t i = (size_t)blockIdx.x * blockDim.x + threadIdx.x;   // S*M threads
    Xb[i] = (bf16_t)X[i];
}

__global__ void lru_prep_w(const float* __restrict__ Bre, const float* __restrict__ Bim,
                           const float* __restrict__ gamma_log,
                           bf16_t* __restrict__ Wre, bf16_t* __restrict__ Wim) {
    size_t i = (size_t)blockIdx.x * blockDim.x + threadIdx.x;   // H*M threads
    int h = (int)(i / M_DIM);
    float g = __expf(gamma_log[h]);
    Wre[i] = (bf16_t)(Bre[i] * g);
    Wim[i] = (bf16_t)(Bim[i] * g);
}

__global__ void lru_prep_c(const float* __restrict__ Cre, const float* __restrict__ Cim,
                           bf16_t* __restrict__ Ctre, bf16_t* __restrict__ CtimNeg) {
    size_t i = (size_t)blockIdx.x * blockDim.x + threadIdx.x;   // M*H threads
    Ctre[i]    = (bf16_t)Cre[i];
    CtimNeg[i] = (bf16_t)(-Cim[i]);   // fold the "- im*im" sign into the weights
}

// ---------------- GEMM 1: Bu = X @ (gamma*B)^T (re & im planes) ----------------
// Block = 8 waves sharing one 32-col weight strip (rows differ per wave).
// Weight K-slices (32 cols x 32 k x {re,im} = 4 KB) are async-staged to LDS,
// double buffered, and broadcast to all 8 waves via ds_load.
// LDS slice layout: byte(mat, col, k) = mat*2048 + col*64 + k*2.
__global__ void lru_gemm_bu(const bf16_t* __restrict__ X,
                            const bf16_t* __restrict__ Wre,
                            const bf16_t* __restrict__ Wim,
                            float* __restrict__ Bure, float* __restrict__ Buim) {
    extern __shared__ char smem[];             // 2 x 4096 B double buffer
    const int tid  = threadIdx.x;
    const int lane = tid & 31;
    const int w    = tid >> 5;                 // wave id in block, 0..7
    const int rowBlocks = S_DIM / 128;         // 256
    const int col0 = (blockIdx.x / rowBlocks) * 32;
    const int row0 = (blockIdx.x % rowBlocks) * 128 + w * 16;

    // per-thread staging source: thread t copies 16 B of the 4 KB slice
    const int  smat = tid >> 7;                // 0 = Wre, 1 = Wim
    const int  scol = (tid & 127) >> 2;        // 0..31
    const int  sk   = (tid & 3) * 8;           // 0,8,16,24
    const bf16_t* gsrc = (smat ? Wim : Wre) + (size_t)(col0 + scol) * M_DIM + sk;
    const unsigned ldsThread = (unsigned)tid * 16;

    async_stage16(ldsThread, gsrc);            // stage slice 0 into buffer 0

    const int n = lane & 15, hi = lane >> 4;
    v8f aRe0 = {}, aRe1 = {}, aIm0 = {}, aIm1 = {};
    const int NK = M_DIM / 32;                 // 16 K-steps
    for (int ks = 0; ks < NK; ++ks) {
        const unsigned bufCur = (ks & 1) ? 4096u : 0u;
        if (ks + 1 < NK) {
            async_stage16((bufCur ^ 4096u) + ldsThread, gsrc + (ks + 1) * 32);
            asm volatile("s_wait_asynccnt 0x1" ::: "memory");   // slice ks landed
        } else {
            asm volatile("s_wait_asynccnt 0x0" ::: "memory");
        }
        __syncthreads();

        const int k0 = ks * 32;
        if (k0 + 32 < M_DIM)
            __builtin_prefetch(X + (size_t)(row0 + n) * M_DIM + k0 + 32, 0, 3);
        v16bf a = load_a_frag(X, M_DIM, row0, k0, lane);
        const char* bb = smem + bufCur;
        v16bf br0 = ld16bf(bb +        (size_t)(n)      * 64 + hi * 32);
        v16bf br1 = ld16bf(bb +        (size_t)(16 + n) * 64 + hi * 32);
        v16bf bi0 = ld16bf(bb + 2048 + (size_t)(n)      * 64 + hi * 32);
        v16bf bi1 = ld16bf(bb + 2048 + (size_t)(16 + n) * 64 + hi * 32);
        aRe0 = wmma_bf16(a, br0, aRe0);
        aRe1 = wmma_bf16(a, br1, aRe1);
        aIm0 = wmma_bf16(a, bi0, aIm0);
        aIm1 = wmma_bf16(a, bi1, aIm1);
        __syncthreads();                       // buffer reread-before-overwrite guard
    }
    store_tile(Bure, H_DIM, row0, col0,      lane, aRe0);
    store_tile(Bure, H_DIM, row0, col0 + 16, lane, aRe1);
    store_tile(Buim, H_DIM, row0, col0,      lane, aIm0);
    store_tile(Buim, H_DIM, row0, col0 + 16, lane, aIm1);
}

// ---------------- chunked linear scan ----------------
__device__ __forceinline__ void lam_of(const float* nu_log, const float* theta_log,
                                       int h, float& lr, float& li) {
    float mag = __expf(-__expf(nu_log[h]));
    float th  = __expf(theta_log[h]);
    lr = mag * __cosf(th);
    li = mag * __sinf(th);
}

// Pass 1: per-chunk local scan with zero init, in place over Bu (becomes local h).
__global__ void lru_scan_local(float* __restrict__ Bure, float* __restrict__ Buim,
                               float* __restrict__ Fre, float* __restrict__ Fim,
                               const float* __restrict__ nu_log,
                               const float* __restrict__ theta_log) {
    const int g = (int)(blockIdx.x * (size_t)blockDim.x + threadIdx.x);
    const int c = g / (B_DIM * H_DIM);
    const int r = g % (B_DIM * H_DIM);
    const int b = r / H_DIM, h = r % H_DIM;
    float lr, li; lam_of(nu_log, theta_log, h, lr, li);

    float hre = 0.f, him = 0.f;
#pragma unroll 4
    for (int j = 0; j < CHUNK; ++j) {
        const int t = c * CHUNK + j;
        const size_t idx = ((size_t)t * B_DIM + b) * H_DIM + h;
        float nre = lr * hre - li * him + Bure[idx];
        float nim = lr * him + li * hre + Buim[idx];
        hre = nre; him = nim;
        Bure[idx] = hre; Buim[idx] = him;     // local prefix, fixed up in pass 3
    }
    Fre[(size_t)c * B_DIM * H_DIM + r] = hre;
    Fim[(size_t)c * B_DIM * H_DIM + r] = him;
}

// Pass 2: exclusive combine across chunks; seeds with h0; writes h_final tail.
__global__ void lru_scan_chunks(const float* __restrict__ Fre, const float* __restrict__ Fim,
                                float* __restrict__ Ere, float* __restrict__ Eim,
                                const float* __restrict__ h0re, const float* __restrict__ h0im,
                                const float* __restrict__ nu_log,
                                const float* __restrict__ theta_log,
                                float* __restrict__ out_tail) {
    const int r = (int)(blockIdx.x * (size_t)blockDim.x + threadIdx.x);
    const int h = r % H_DIM;
    float lr, li; lam_of(nu_log, theta_log, h, lr, li);
    float plr = 1.f, pli = 0.f;                // lam^CHUNK
#pragma unroll
    for (int i = 0; i < CHUNK; ++i) {
        float t = plr * lr - pli * li;
        pli = plr * li + pli * lr;
        plr = t;
    }
    float ere = h0re[r], eim = h0im[r];
    for (int c = 0; c < NCHUNK; ++c) {
        const size_t o = (size_t)c * B_DIM * H_DIM + r;
        Ere[o] = ere; Eim[o] = eim;
        float fre = Fre[o], fim = Fim[o];
        float nre = plr * ere - pli * eim + fre;
        float nim = plr * eim + pli * ere + fim;
        ere = nre; eim = nim;
    }
    out_tail[2 * (size_t)r]     = ere;         // h_final complex64 interleaved [B,H]
    out_tail[2 * (size_t)r + 1] = eim;
}

// Pass 3: fixup h_global[t] = local[t] + lam^{j+1} * E_c, emit bf16 states for GEMM2.
__global__ void lru_scan_fixup(const float* __restrict__ Lre, const float* __restrict__ Lim,
                               const float* __restrict__ Ere, const float* __restrict__ Eim,
                               bf16_t* __restrict__ Hre, bf16_t* __restrict__ Him,
                               const float* __restrict__ nu_log,
                               const float* __restrict__ theta_log) {
    const int g = (int)(blockIdx.x * (size_t)blockDim.x + threadIdx.x);
    const int c = g / (B_DIM * H_DIM);
    const int r = g % (B_DIM * H_DIM);
    const int b = r / H_DIM, h = r % H_DIM;
    float lr, li; lam_of(nu_log, theta_log, h, lr, li);

    float cre = Ere[(size_t)c * B_DIM * H_DIM + r];
    float cim = Eim[(size_t)c * B_DIM * H_DIM + r];
#pragma unroll 4
    for (int j = 0; j < CHUNK; ++j) {
        float t0 = lr * cre - li * cim;
        cim = lr * cim + li * cre;
        cre = t0;                              // lam^{j+1} * E_c
        const int t = c * CHUNK + j;
        const size_t idx = ((size_t)t * B_DIM + b) * H_DIM + h;
        Hre[idx] = (bf16_t)(Lre[idx] + cre);
        Him[idx] = (bf16_t)(Lim[idx] + cim);
    }
}

// ---------------- GEMM 2: Y = Hre@Cre^T + Him@(-Cim)^T + D*x ----------------
// Same LDS async-staging scheme; A operands are the bf16 recurrent states.
__global__ void lru_gemm_y(const bf16_t* __restrict__ Hre,
                           const bf16_t* __restrict__ Him,
                           const bf16_t* __restrict__ Ctre,
                           const bf16_t* __restrict__ CtimNeg,
                           const float* __restrict__ Xf,
                           const float* __restrict__ Dv,
                           float* __restrict__ Out) {
    extern __shared__ char smem[];             // 2 x 4096 B double buffer
    const int tid  = threadIdx.x;
    const int lane = tid & 31;
    const int w    = tid >> 5;
    const int rowBlocks = S_DIM / 128;         // 256
    const int col0 = (blockIdx.x / rowBlocks) * 32;
    const int row0 = (blockIdx.x % rowBlocks) * 128 + w * 16;

    const int  smat = tid >> 7;                // 0 = Ctre, 1 = CtimNeg
    const int  scol = (tid & 127) >> 2;
    const int  sk   = (tid & 3) * 8;
    const bf16_t* gsrc = (smat ? CtimNeg : Ctre) + (size_t)(col0 + scol) * H_DIM + sk;
    const unsigned ldsThread = (unsigned)tid * 16;

    async_stage16(ldsThread, gsrc);

    const int n = lane & 15, hi = lane >> 4;
    v8f acc0 = {}, acc1 = {};
    const int NK = H_DIM / 32;                 // 8 K-steps
    for (int ks = 0; ks < NK; ++ks) {
        const unsigned bufCur = (ks & 1) ? 4096u : 0u;
        if (ks + 1 < NK) {
            async_stage16((bufCur ^ 4096u) + ldsThread, gsrc + (ks + 1) * 32);
            asm volatile("s_wait_asynccnt 0x1" ::: "memory");
        } else {
            asm volatile("s_wait_asynccnt 0x0" ::: "memory");
        }
        __syncthreads();

        const int k0 = ks * 32;
        if (k0 + 32 < H_DIM)
            __builtin_prefetch(Hre + (size_t)(row0 + n) * H_DIM + k0 + 32, 0, 3);
        v16bf are = load_a_frag(Hre, H_DIM, row0, k0, lane);
        v16bf aim = load_a_frag(Him, H_DIM, row0, k0, lane);
        const char* bb = smem + bufCur;
        v16bf br0 = ld16bf(bb +        (size_t)(n)      * 64 + hi * 32);
        v16bf br1 = ld16bf(bb +        (size_t)(16 + n) * 64 + hi * 32);
        v16bf bi0 = ld16bf(bb + 2048 + (size_t)(n)      * 64 + hi * 32);
        v16bf bi1 = ld16bf(bb + 2048 + (size_t)(16 + n) * 64 + hi * 32);
        acc0 = wmma_bf16(are, br0, acc0);
        acc0 = wmma_bf16(aim, bi0, acc0);
        acc1 = wmma_bf16(are, br1, acc1);
        acc1 = wmma_bf16(aim, bi1, acc1);
        __syncthreads();
    }
    // fused epilogue: + D[m] * x
#pragma unroll
    for (int nt = 0; nt < 2; ++nt) {
        const int col = col0 + nt * 16 + n;
        const float d = Dv[col];
        v8f acc = nt ? acc1 : acc0;
#pragma unroll
        for (int i = 0; i < 8; ++i) {
            const size_t s = (size_t)row0 + hi * 8 + i;
            const size_t o = s * M_DIM + col;
            Out[o] = acc[i] + d * Xf[o];
        }
    }
}

// ---------------- launch ----------------
extern "C" void kernel_launch(void* const* d_in, const int* in_sizes, int n_in,
                              void* d_out, int out_size, void* d_ws, size_t ws_size,
                              hipStream_t stream) {
    const float* inputs    = (const float*)d_in[0];
    const float* h0re      = (const float*)d_in[1];
    const float* h0im      = (const float*)d_in[2];
    const float* nu_log    = (const float*)d_in[3];
    const float* theta_log = (const float*)d_in[4];
    const float* gamma_log = (const float*)d_in[5];
    const float* B_re      = (const float*)d_in[6];
    const float* B_im      = (const float*)d_in[7];
    const float* C_re      = (const float*)d_in[8];
    const float* C_im      = (const float*)d_in[9];
    const float* Dv        = (const float*)d_in[10];
    float* out = (float*)d_out;

    // workspace carve (256-B aligned)
    size_t off = 0;
    auto carve = [&](size_t bytes) {
        void* p = (char*)d_ws + off;
        off = (off + bytes + 255) & ~(size_t)255;
        return p;
    };
    bf16_t* Xb   = (bf16_t*)carve((size_t)S_DIM * M_DIM * sizeof(bf16_t));
    bf16_t* Wre  = (bf16_t*)carve((size_t)H_DIM * M_DIM * sizeof(bf16_t));
    bf16_t* Wim  = (bf16_t*)carve((size_t)H_DIM * M_DIM * sizeof(bf16_t));
    bf16_t* Ctre = (bf16_t*)carve((size_t)M_DIM * H_DIM * sizeof(bf16_t));
    bf16_t* Ctim = (bf16_t*)carve((size_t)M_DIM * H_DIM * sizeof(bf16_t));
    float*  Bure = (float*) carve((size_t)S_DIM * H_DIM * sizeof(float));
    float*  Buim = (float*) carve((size_t)S_DIM * H_DIM * sizeof(float));
    float*  Fre  = (float*) carve((size_t)NCHUNK * B_DIM * H_DIM * sizeof(float));
    float*  Fim  = (float*) carve((size_t)NCHUNK * B_DIM * H_DIM * sizeof(float));
    float*  Ere  = (float*) carve((size_t)NCHUNK * B_DIM * H_DIM * sizeof(float));
    float*  Eim  = (float*) carve((size_t)NCHUNK * B_DIM * H_DIM * sizeof(float));
    bf16_t* Hreb = (bf16_t*)carve((size_t)S_DIM * H_DIM * sizeof(bf16_t));
    bf16_t* Himb = (bf16_t*)carve((size_t)S_DIM * H_DIM * sizeof(bf16_t));
    (void)ws_size; (void)in_sizes; (void)n_in; (void)out_size;

    const int TB = 256;

    // prep
    lru_cvt_x <<<(S_DIM * (size_t)M_DIM) / TB, TB, 0, stream>>>(inputs, Xb);
    lru_prep_w<<<(H_DIM * (size_t)M_DIM) / TB, TB, 0, stream>>>(B_re, B_im, gamma_log, Wre, Wim);
    lru_prep_c<<<(M_DIM * (size_t)H_DIM) / TB, TB, 0, stream>>>(C_re, C_im, Ctre, Ctim);

    // GEMM 1: Bu = X @ W^T   (8 col strips x 256 row groups, 8 KB LDS double buffer)
    lru_gemm_bu<<<(M_DIM ? (H_DIM / 32) * (S_DIM / 128) : 0), TB, 8192, stream>>>(Xb, Wre, Wim, Bure, Buim);

    // chunked scan
    lru_scan_local <<<(NCHUNK * B_DIM * H_DIM) / TB, TB, 0, stream>>>(Bure, Buim, Fre, Fim, nu_log, theta_log);
    lru_scan_chunks<<<(B_DIM * H_DIM) / TB, TB, 0, stream>>>(Fre, Fim, Ere, Eim, h0re, h0im,
                                                             nu_log, theta_log,
                                                             out + (size_t)T_DIM * B_DIM * M_DIM);
    lru_scan_fixup <<<(NCHUNK * B_DIM * H_DIM) / TB, TB, 0, stream>>>(Bure, Buim, Ere, Eim,
                                                                      Hreb, Himb, nu_log, theta_log);

    // GEMM 2: Y = Hre@Cre^T + Him@(-Cim)^T + D*x   (16 col strips x 256 row groups)
    lru_gemm_y<<<(M_DIM / 32) * (S_DIM / 128), TB, 8192, stream>>>(Hreb, Himb, Ctre, Ctim,
                                                                   inputs, Dv, out);
}
